// GraphRegressorCond_12704513261988
// MI455X (gfx1250) — matrix operand order repaired and checked by reading, not compiled, not run on hardware
//
#include <hip/hip_runtime.h>

// ---------------------------------------------------------------- types
typedef __bf16 bf16;
typedef __attribute__((ext_vector_type(16))) __bf16 bf16x16;
typedef __attribute__((ext_vector_type(8)))  __bf16 bf16x8;
typedef __attribute__((ext_vector_type(8)))  float  f32x8;

static constexpr int N_NODES  = 50000;
static constexpr int N_EDGES  = 800000;
static constexpr int N_GRAPHS = 128;
static constexpr int IN_CH    = 16;
static constexpr int IN_CTX   = 16;
static constexpr int HDIM     = 128;   // H_G = H_C = H_F

__device__ __forceinline__ bf16 to_bf16(float f) { return (bf16)f; }

// ---------------------------------------------------------------- utility
__global__ void k_fill_f32(float* p, float v, int n) {
    int t = blockIdx.x * blockDim.x + threadIdx.x;
    if (t < n) p[t] = v;
}

// deg[dst] += ew  (deg pre-filled with 1.0 = self-loop weight)
__global__ void k_deg(const long long* __restrict__ ei,
                      const float* __restrict__ ew,
                      float* __restrict__ deg, int E) {
    int e = blockIdx.x * blockDim.x + threadIdx.x;
    if (e < E) atomicAdd(&deg[(int)ei[(long)E + e]], ew[e]);
}

// dinv = rsqrt(deg); dinv2 = 1/deg  (deg >= 1 always: self loop)
__global__ void k_dinv(const float* __restrict__ deg,
                       float* __restrict__ dinv, float* __restrict__ dinv2, int n) {
    int t = blockIdx.x * blockDim.x + threadIdx.x;
    if (t < n) { float r = rsqrtf(deg[t]); dinv[t] = r; dinv2[t] = r * r; }
}

__global__ void k_norm(const long long* __restrict__ ei,
                       const float* __restrict__ ew,
                       const float* __restrict__ dinv,
                       float* __restrict__ nrm, int E) {
    int e = blockIdx.x * blockDim.x + threadIdx.x;
    if (e < E) nrm[e] = dinv[(int)ei[e]] * ew[e] * dinv[(int)ei[(long)E + e]];
}

// convert f32 [rows x Kin] -> bf16 [rows x Kpad], zero-padded columns
__global__ void k_convert_pad(const float* __restrict__ X, bf16* __restrict__ Y,
                              int Kin, int Kpad, int total) {
    int t = blockIdx.x * blockDim.x + threadIdx.x;
    if (t >= total) return;
    int r = t / Kpad, c = t - r * Kpad;
    Y[t] = (c < Kin) ? to_bf16(X[r * Kin + c]) : to_bf16(0.0f);
}

// ---------------------------------------------------------------- weight packing
// Pack W f32 [K x 128] into B-fragment layout for v_wmma_f32_16x16x32_bf16:
//   pk[((ks*8 + nt)*32 + lane)*16 + j]  with  k = ks*32 + (lane>>4)*16 + j,
//   n = nt*16 + (lane&15).   (this code's k order, same as load_a_frag: lanes 0-15 K=0..15, lanes 16-31 K=16..31)
__global__ void k_pack_w(const float* __restrict__ W, bf16* __restrict__ pk,
                         int K, int total) {
    int t = blockIdx.x * blockDim.x + threadIdx.x;
    if (t >= total) return;
    int j  =  t        & 15;
    int l  = (t >> 4)  & 31;
    int nt = (t >> 9)  & 7;
    int ks =  t >> 12;
    int k  = ks * 32 + (l >> 4) * 16 + j;
    int n  = nt * 16 + (l & 15);
    pk[t] = (k < K) ? to_bf16(W[k * HDIM + n]) : to_bf16(0.0f);
}

// ---------------------------------------------------------------- WMMA fragments
// A (16-bit 16x32): lane l holds row m=l&15; K runs {16h..16h+15}, h=l>>4
__device__ __forceinline__ bf16x16 load_a_frag(const bf16* __restrict__ arow,
                                               int ks, int half) {
    bf16x8 a0 = *(const bf16x8*)(arow + ks * 32 + half * 16);
    bf16x8 a1 = *(const bf16x8*)(arow + ks * 32 + half * 16 + 8);
    return __builtin_shufflevector(a0, a1, 0,1,2,3,4,5,6,7,8,9,10,11,12,13,14,15);
}

// ---------------------------------------------------------------- GCN GEMM
// Two M-tiles per wave: independent WMMA chains (no RAW hazard NOPs) + B reuse.
// hw = A @ W ;  out = dinv2[m] * hw   (self-loop term == scatter init)
__global__ void k_gemm_gcn(const bf16* __restrict__ A, int strideA, int Ksteps,
                           const bf16* __restrict__ Bpk,
                           const float* __restrict__ dinv2,
                           float* __restrict__ hw, float* __restrict__ outacc,
                           int Mtiles, int Mpairs) {
    int pair = blockIdx.x * blockDim.y + threadIdx.y;
    if (pair >= Mpairs) return;                 // wave-uniform
    int t0 = pair * 2;
    int nt = blockIdx.y;
    int l = threadIdx.x, half = l >> 4, lo = l & 15;
    const bf16* arow0 = A + (long)(t0 * 16 + lo) * strideA;
    const bf16* arow1 = arow0 + (long)16 * strideA;   // padded: always in-bounds

    f32x8 c0 = {}, c1 = {};
    for (int ks = 0; ks < Ksteps; ++ks) {
        bf16x16 a0 = load_a_frag(arow0, ks, half);
        bf16x16 a1 = load_a_frag(arow1, ks, half);
        bf16x16 b  = *(const bf16x16*)(Bpk + ((long)(ks * 8 + nt) * 32 + l) * 16);
        c0 = __builtin_amdgcn_wmma_f32_16x16x32_bf16(false, a0, false, b,
                                                     (short)0, c0, false, false);
        c1 = __builtin_amdgcn_wmma_f32_16x16x32_bf16(false, a1, false, b,
                                                     (short)0, c1, false, false);
    }
    int col = nt * 16 + lo;
    int mb0 = t0 * 16 + half * 8;               // C/D: VGPR r -> M = 8*half + r
#pragma unroll
    for (int r = 0; r < 8; ++r) {
        int m = mb0 + r;
        float v = c0[r];
        int idx = m * HDIM + col;
        hw[idx] = v;
        outacc[idx] = dinv2[m] * v;
    }
    if (t0 + 1 < Mtiles) {                      // wave-uniform guard
#pragma unroll
        for (int r = 0; r < 8; ++r) {
            int m = mb0 + 16 + r;
            float v = c1[r];
            int idx = m * HDIM + col;
            hw[idx] = v;
            outacc[idx] = dinv2[m] * v;
        }
    }
}

// ---------------------------------------------------------------- edge scatter
// out[dst] += norm[e] * hw[src]; 32 lanes per edge pair, float4 per lane,
// prefetch second edge's gather row to overlap with first edge's atomics.
__global__ void k_scatter(const long long* __restrict__ ei,
                          const float* __restrict__ nrm,
                          const float* __restrict__ hw,
                          float* __restrict__ outacc, int E) {
    int t = blockIdx.x * blockDim.x + threadIdx.x;
    int p = t >> 5;
    long e0 = (long)p * 2;
    if (e0 >= E) return;
    int lane4 = (t & 31) * 4;

    int src0 = (int)ei[e0], dst0 = (int)ei[(long)E + e0];
    float w0 = nrm[e0];
    bool has1 = (e0 + 1) < E;                   // wave-uniform
    int src1 = 0, dst1 = 0; float w1 = 0.0f;
    if (has1) {
        src1 = (int)ei[e0 + 1];
        dst1 = (int)ei[(long)E + e0 + 1];
        w1   = nrm[e0 + 1];
        __builtin_prefetch(hw + (long)src1 * HDIM + lane4, 0, 0);
    }
    float4 v0 = *(const float4*)(hw + (long)src0 * HDIM + lane4);
    float* o0 = outacc + (long)dst0 * HDIM + lane4;
    atomicAdd(o0 + 0, w0 * v0.x);
    atomicAdd(o0 + 1, w0 * v0.y);
    atomicAdd(o0 + 2, w0 * v0.z);
    atomicAdd(o0 + 3, w0 * v0.w);
    if (has1) {
        float4 v1 = *(const float4*)(hw + (long)src1 * HDIM + lane4);
        float* o1 = outacc + (long)dst1 * HDIM + lane4;
        atomicAdd(o1 + 0, w1 * v1.x);
        atomicAdd(o1 + 1, w1 * v1.y);
        atomicAdd(o1 + 2, w1 * v1.z);
        atomicAdd(o1 + 3, w1 * v1.w);
    }
}

// h = relu(out + b) -> bf16 (next layer A); optional f32 copy (for pooling)
__global__ void k_finalize(const float* __restrict__ outacc,
                           const float* __restrict__ bias,
                           bf16* __restrict__ hb, float* __restrict__ hf, int total) {
    int t = blockIdx.x * blockDim.x + threadIdx.x;
    if (t >= total) return;
    float v = fmaxf(outacc[t] + bias[t & (HDIM - 1)], 0.0f);
    hb[t] = to_bf16(v);
    if (hf) hf[t] = v;
}

// ---------------------------------------------------------------- pooling
__global__ void k_pool_sum(const float* __restrict__ hf,
                           const long long* __restrict__ batch,
                           float* __restrict__ s, float* __restrict__ cnt, int total) {
    int t = blockIdx.x * blockDim.x + threadIdx.x;
    if (t >= total) return;
    int node = t >> 7, col = t & (HDIM - 1);
    int g = (int)batch[node];
    atomicAdd(&s[g * HDIM + col], hf[t]);
    if (col == 0) atomicAdd(&cnt[g], 1.0f);
}

// z[:, 0:128] = bf16(s / max(cnt,1))  (row stride 256 for concat)
__global__ void k_pool_div(const float* __restrict__ s, const float* __restrict__ cnt,
                           bf16* __restrict__ z, int total) {
    int t = blockIdx.x * blockDim.x + threadIdx.x;
    if (t >= total) return;
    int g = t >> 7, col = t & (HDIM - 1);
    z[g * 256 + col] = to_bf16(s[t] / fmaxf(cnt[g], 1.0f));
}

// ---------------------------------------------------------------- MLP GEMM
// D = relu?(A@W + b); two M-tiles per wave; store bf16 (stride/colOff) and/or f32
__global__ void k_gemm_mlp(const bf16* __restrict__ A, int strideA, int Ksteps,
                           const bf16* __restrict__ Bpk,
                           const float* __restrict__ bias, int do_relu,
                           bf16* __restrict__ outb, int strideOut, int colOff,
                           float* __restrict__ outf, int Mtiles, int Mpairs) {
    int pair = blockIdx.x * blockDim.y + threadIdx.y;
    if (pair >= Mpairs) return;
    int t0 = pair * 2;
    int nt = blockIdx.y;
    int l = threadIdx.x, half = l >> 4, lo = l & 15;
    const bf16* arow0 = A + (long)(t0 * 16 + lo) * strideA;
    const bf16* arow1 = arow0 + (long)16 * strideA;

    f32x8 c0 = {}, c1 = {};
    for (int ks = 0; ks < Ksteps; ++ks) {
        bf16x16 a0 = load_a_frag(arow0, ks, half);
        bf16x16 a1 = load_a_frag(arow1, ks, half);
        bf16x16 b  = *(const bf16x16*)(Bpk + ((long)(ks * 8 + nt) * 32 + l) * 16);
        c0 = __builtin_amdgcn_wmma_f32_16x16x32_bf16(false, a0, false, b,
                                                     (short)0, c0, false, false);
        c1 = __builtin_amdgcn_wmma_f32_16x16x32_bf16(false, a1, false, b,
                                                     (short)0, c1, false, false);
    }
    int col = nt * 16 + lo;
    float bcol = bias[col];
    for (int tt = 0; tt < 2; ++tt) {
        if (t0 + tt >= Mtiles) break;           // wave-uniform
        int mb = (t0 + tt) * 16 + half * 8;
#pragma unroll
        for (int r = 0; r < 8; ++r) {
            int m = mb + r;
            float v = (tt ? c1[r] : c0[r]) + bcol;
            if (do_relu) v = fmaxf(v, 0.0f);
            if (outb) outb[m * strideOut + colOff + col] = to_bf16(v);
            if (outf) outf[m * HDIM + col] = v;
        }
    }
}

// ---------------------------------------------------------------- host
extern "C" void kernel_launch(void* const* d_in, const int* in_sizes, int n_in,
                              void* d_out, int out_size, void* d_ws, size_t ws_size,
                              hipStream_t stream) {
    (void)in_sizes; (void)n_in; (void)out_size; (void)ws_size;

    const float*     x     = (const float*)d_in[0];
    const float*     xc    = (const float*)d_in[1];
    const long long* ei    = (const long long*)d_in[2];   // int64 per reference
    const long long* batch = (const long long*)d_in[3];
    const float*     ew    = (const float*)d_in[4];
    const float* gW[4] = {(const float*)d_in[5],  (const float*)d_in[7],
                          (const float*)d_in[9],  (const float*)d_in[11]};
    const float* gb[4] = {(const float*)d_in[6],  (const float*)d_in[8],
                          (const float*)d_in[10], (const float*)d_in[12]};
    const float* cW[3] = {(const float*)d_in[13], (const float*)d_in[15],
                          (const float*)d_in[17]};
    const float* cb[3] = {(const float*)d_in[14], (const float*)d_in[16],
                          (const float*)d_in[18]};
    const float* fW[3] = {(const float*)d_in[19], (const float*)d_in[21],
                          (const float*)d_in[23]};
    const float* fb[3] = {(const float*)d_in[20], (const float*)d_in[22],
                          (const float*)d_in[24]};

    // ---- bump allocator over workspace
    char* ws = (char*)d_ws;
    size_t off = 0;
    auto alloc = [&](size_t bytes) -> void* {
        off = (off + 255) & ~(size_t)255;
        void* p = ws + off;
        off += bytes;
        return p;
    };
    const int NPAD = N_NODES + 16;              // 16 pad rows for phantom M-tile
    float* deg    = (float*)alloc((size_t)N_NODES * 4);
    float* dinv   = (float*)alloc((size_t)N_NODES * 4);
    float* dinv2  = (float*)alloc((size_t)N_NODES * 4);
    float* nrm    = (float*)alloc((size_t)N_EDGES * 4);
    bf16*  x_bf   = (bf16*) alloc((size_t)NPAD * 32 * 2);
    bf16*  h_bf   = (bf16*) alloc((size_t)NPAD * HDIM * 2);
    float* hw     = (float*)alloc((size_t)N_NODES * HDIM * 4);
    float* outacc = (float*)alloc((size_t)N_NODES * HDIM * 4);
    float* pool_s = (float*)alloc((size_t)N_GRAPHS * HDIM * 4);
    float* cnt    = (float*)alloc((size_t)N_GRAPHS * 4);
    bf16*  ctx_bf = (bf16*) alloc((size_t)N_GRAPHS * 32 * 2);
    bf16*  cbufA  = (bf16*) alloc((size_t)N_GRAPHS * HDIM * 2);
    bf16*  cbufB  = (bf16*) alloc((size_t)N_GRAPHS * HDIM * 2);
    bf16*  z_bf   = (bf16*) alloc((size_t)N_GRAPHS * 256 * 2);
    bf16*  zb1    = (bf16*) alloc((size_t)N_GRAPHS * HDIM * 2);
    bf16*  zb2    = (bf16*) alloc((size_t)N_GRAPHS * HDIM * 2);

    auto pk_bytes = [](int Ksteps) { return (size_t)Ksteps * 8 * 32 * 16 * 2; };
    bf16* pk_g[4]; int gK[4] = {IN_CH, HDIM, HDIM, HDIM};
    bf16* pk_c[3]; int cK[3] = {IN_CTX, HDIM, HDIM};
    bf16* pk_f[3]; int fK[3] = {2 * HDIM, HDIM, HDIM};
    for (int i = 0; i < 4; ++i) pk_g[i] = (bf16*)alloc(pk_bytes((gK[i] + 31) / 32));
    for (int i = 0; i < 3; ++i) pk_c[i] = (bf16*)alloc(pk_bytes((cK[i] + 31) / 32));
    for (int i = 0; i < 3; ++i) pk_f[i] = (bf16*)alloc(pk_bytes((fK[i] + 31) / 32));

    const int B = 256;
    auto nb = [&](long n) { return (unsigned)((n + B - 1) / B); };

    // ---- precompute: degree / norm ------------------------------------
    k_fill_f32<<<nb(N_NODES), B, 0, stream>>>(deg, 1.0f, N_NODES);
    k_fill_f32<<<nb(N_GRAPHS * HDIM), B, 0, stream>>>(pool_s, 0.0f, N_GRAPHS * HDIM);
    k_fill_f32<<<nb(N_GRAPHS), B, 0, stream>>>(cnt, 0.0f, N_GRAPHS);
    k_deg <<<nb(N_EDGES), B, 0, stream>>>(ei, ew, deg, N_EDGES);
    k_dinv<<<nb(N_NODES), B, 0, stream>>>(deg, dinv, dinv2, N_NODES);
    k_norm<<<nb(N_EDGES), B, 0, stream>>>(ei, ew, dinv, nrm, N_EDGES);

    // ---- pack all weights into WMMA B-fragment layout -----------------
    for (int i = 0; i < 4; ++i) {
        int tot = ((gK[i] + 31) / 32) * 4096;
        k_pack_w<<<nb(tot), B, 0, stream>>>(gW[i], pk_g[i], gK[i], tot);
    }
    for (int i = 0; i < 3; ++i) {
        int tot = ((cK[i] + 31) / 32) * 4096;
        k_pack_w<<<nb(tot), B, 0, stream>>>(cW[i], pk_c[i], cK[i], tot);
    }
    for (int i = 0; i < 3; ++i) {
        int tot = ((fK[i] + 31) / 32) * 4096;
        k_pack_w<<<nb(tot), B, 0, stream>>>(fW[i], pk_f[i], fK[i], tot);
    }

    // ---- activations to bf16 ------------------------------------------
    k_convert_pad<<<nb((long)N_NODES * 32), B, 0, stream>>>(x, x_bf, IN_CH, 32,
                                                            N_NODES * 32);
    k_convert_pad<<<nb(N_GRAPHS * 32), B, 0, stream>>>(xc, ctx_bf, IN_CTX, 32,
                                                       N_GRAPHS * 32);

    // ---- 4 GCN layers --------------------------------------------------
    const int MT  = N_NODES / 16;               // 3125
    const int MP  = (MT + 1) / 2;               // 1563 tile-pairs
    dim3 gemm_blk(32, 4);
    dim3 gcn_grid((MP + 3) / 4, 8);
    const bf16* Ain = x_bf; int strideA = 32, Ksteps = 1;
    for (int L = 0; L < 4; ++L) {
        k_gemm_gcn<<<gcn_grid, gemm_blk, 0, stream>>>(Ain, strideA, Ksteps, pk_g[L],
                                                      dinv2, hw, outacc, MT, MP);
        k_scatter<<<nb((long)((N_EDGES + 1) / 2) * 32), B, 0, stream>>>(
            ei, nrm, hw, outacc, N_EDGES);
        k_finalize<<<nb((long)N_NODES * HDIM), B, 0, stream>>>(
            outacc, gb[L], h_bf, (L == 3) ? hw : nullptr, N_NODES * HDIM);
        Ain = h_bf; strideA = HDIM; Ksteps = 4;
    }

    // ---- global mean pool -> z[:, 0:128] -------------------------------
    k_pool_sum<<<nb((long)N_NODES * HDIM), B, 0, stream>>>(hw, batch, pool_s, cnt,
                                                           N_NODES * HDIM);
    k_pool_div<<<nb(N_GRAPHS * HDIM), B, 0, stream>>>(pool_s, cnt, z_bf,
                                                      N_GRAPHS * HDIM);

    // ---- context MLP -> z[:, 128:256] ----------------------------------
    const int MTg = N_GRAPHS / 16;              // 8
    const int MPg = MTg / 2;                    // 4
    dim3 mlp_grid((MPg + 3) / 4, 8);
    k_gemm_mlp<<<mlp_grid, gemm_blk, 0, stream>>>(ctx_bf, 32, 1, pk_c[0], cb[0], 1,
                                                  cbufA, HDIM, 0, nullptr, MTg, MPg);
    k_gemm_mlp<<<mlp_grid, gemm_blk, 0, stream>>>(cbufA, HDIM, 4, pk_c[1], cb[1], 1,
                                                  cbufB, HDIM, 0, nullptr, MTg, MPg);
    k_gemm_mlp<<<mlp_grid, gemm_blk, 0, stream>>>(cbufB, HDIM, 4, pk_c[2], cb[2], 1,
                                                  z_bf, 256, HDIM, nullptr, MTg, MPg);

    // ---- FC head --------------------------------------------------------
    k_gemm_mlp<<<mlp_grid, gemm_blk, 0, stream>>>(z_bf, 256, 8, pk_f[0], fb[0], 1,
                                                  zb1, HDIM, 0, nullptr, MTg, MPg);
    k_gemm_mlp<<<mlp_grid, gemm_blk, 0, stream>>>(zb1, HDIM, 4, pk_f[1], fb[1], 1,
                                                  zb2, HDIM, 0, nullptr, MTg, MPg);
    k_gemm_mlp<<<mlp_grid, gemm_blk, 0, stream>>>(zb2, HDIM, 4, pk_f[2], fb[2], 0,
                                                  nullptr, 0, 0, (float*)d_out, MTg, MPg);
}
